// CustomRNN_944892805527
// MI455X (gfx1250) — compile-verified, matrix operand order checked by reading, not run on hardware
//
#include <hip/hip_runtime.h>
#include <math.h>

// Problem constants (from reference)
#define ALPHA_F 0.1f
#define B_SZ 64
#define T_SZ 1024
#define I_SZ 256
#define H_SZ 512
#define O_SZ 256
#define NWG_SCAN 8   // persistent scan workgroups (each owns 64 H-columns)

typedef __attribute__((ext_vector_type(16))) __bf16        v16bf;
typedef __attribute__((ext_vector_type(8)))  float         v8f;
typedef __attribute__((ext_vector_type(4)))  unsigned int  u32x4;

// f32 -> bf16 with round-to-nearest-even
__device__ __forceinline__ unsigned short f2bf(float f) {
    unsigned int u = __float_as_uint(f);
    u += 0x7FFFu + ((u >> 16) & 1u);
    return (unsigned short)(u >> 16);
}
__device__ __forceinline__ unsigned int pk2bf(float a, float b) {
    return (unsigned int)f2bf(a) | ((unsigned int)f2bf(b) << 16);
}

// CDNA5 async global->LDS copy (ASYNCcnt-tracked). lds_off = low 32 bits of a
// flat pointer into LDS (== LDS byte offset per the aperture mapping).
__device__ __forceinline__ void async_ld_b128(void* lds_ptr, const void* gptr) {
    unsigned int lds_off = (unsigned int)(uintptr_t)lds_ptr;
    unsigned long long ga = (unsigned long long)(uintptr_t)gptr;
    asm volatile("global_load_async_to_lds_b128 %0, %1, off"
                 :: "v"(lds_off), "v"(ga) : "memory");
}
__device__ __forceinline__ void wait_async0() {
    asm volatile("s_wait_asynccnt 0x0" ::: "memory");
}

// Fragment addressing per CDNA5 ISA (16x32 bf16 tile, 16 bf16/lane).
__device__ __forceinline__ const unsigned short*
frag_addr(const unsigned short* base, int ldk, int row0, int k0, int lane) {
    int r = lane & 15, half = lane >> 4;
    return base + (row0 + r) * ldk + k0 + half * 8;
}
__device__ __forceinline__ v16bf frag_ld(const unsigned short* p) {
    union { u32x4 u[2]; v16bf b; } f;
    f.u[0] = *(const u32x4*)(p);
    f.u[1] = *(const u32x4*)(p + 16);
    return f.b;
}
__device__ __forceinline__ v8f wmma_bf16(v16bf a, v16bf b, v8f c) {
    return __builtin_amdgcn_wmma_f32_16x16x32_bf16(false, a, false, b,
                                                   (short)0, c, false, false);
}

// Staging helpers for the GEMM kernels (256 threads).
__device__ __forceinline__ void
stage_A64x32(unsigned short* As, const float* __restrict__ src, int ld,
             int row0, int k0, int tid) {
    int i = tid * 8;                      // 2048 elems = 256 thr * 8
    int m = i >> 5, k = i & 31;
    const float4* p = (const float4*)&src[(size_t)(row0 + m) * ld + k0 + k];
    float4 v0 = p[0], v1 = p[1];
    u32x4 pk = { pk2bf(v0.x, v0.y), pk2bf(v0.z, v0.w),
                 pk2bf(v1.x, v1.y), pk2bf(v1.z, v1.w) };
    *(u32x4*)&As[i] = pk;
}
__device__ __forceinline__ void
stage_B128x32(unsigned short* Bs, const float* __restrict__ src, int ld,
              int k0, int n0, int tid) {
    #pragma unroll
    for (int i = tid * 4; i < 32 * 128; i += 256 * 4) {   // transpose stage
        int k = i >> 7, n = i & 127;
        float4 v = *(const float4*)&src[(size_t)(k0 + k) * ld + n0 + n];
        Bs[(n + 0) * 32 + k] = f2bf(v.x);
        Bs[(n + 1) * 32 + k] = f2bf(v.y);
        Bs[(n + 2) * 32 + k] = f2bf(v.z);
        Bs[(n + 3) * 32 + k] = f2bf(v.w);
    }
}

// ---------------------------------------------------------------------------
// Prep: W_recT_bf16[n][k] = bf16(W_rec[k][n]) ; hbf0 = bf16(h0)
// ---------------------------------------------------------------------------
__global__ void __launch_bounds__(256)
k_prep_wrec(const float* __restrict__ Wrec, unsigned short* __restrict__ WrecT) {
    int i = blockIdx.x * 256 + threadIdx.x;      // i < H*H
    int n = i >> 9, k = i & (H_SZ - 1);
    WrecT[i] = f2bf(Wrec[(size_t)k * H_SZ + n]);
}
__global__ void __launch_bounds__(256)
k_prep_h(const float* __restrict__ h0, unsigned short* __restrict__ hbf0) {
    int i = blockIdx.x * 256 + threadIdx.x;      // i < B*H
    hbf0[i] = f2bf(h0[i]);
}

// ---------------------------------------------------------------------------
// Kernel 1: xin = x @ W_in, stored as [T,B,H] f32 for the scan.
// 256 thr (8 waves), tile 64(M) x 128(N); double-buffered LDS staging.
// ---------------------------------------------------------------------------
__global__ void __launch_bounds__(256)
k_in_proj(const float* __restrict__ x, const float* __restrict__ Win,
          float* __restrict__ xin) {
    __shared__ __align__(16) unsigned short As[2][64 * 32];
    __shared__ __align__(16) unsigned short Bs[2][128 * 32];
    const int tid  = threadIdx.x;
    const int row0 = blockIdx.x * 64;
    const int n0   = blockIdx.y * 128;
    const int w = tid >> 5, lane = tid & 31;
    const int mi = w & 3, ng = w >> 2;          // 4 M-tiles x 2 N-groups
    const int half = lane >> 4, nl = lane & 15;
    const int NKB = I_SZ / 32;

    stage_A64x32(As[0], x, I_SZ, row0, 0, tid);
    stage_B128x32(Bs[0], Win, H_SZ, 0, n0, tid);
    __syncthreads();

    v8f c[4] = {};
    for (int kb = 0; kb < NKB; ++kb) {
        int cur = kb & 1;
        if (kb + 1 < NKB) {                      // overlap next stage w/ WMMA
            stage_A64x32(As[cur ^ 1], x, I_SZ, row0, (kb + 1) * 32, tid);
            stage_B128x32(Bs[cur ^ 1], Win, H_SZ, (kb + 1) * 32, n0, tid);
        }
        v16bf a = frag_ld(frag_addr(As[cur], 32, mi * 16, 0, lane));
        v16bf bf4[4];
        #pragma unroll
        for (int j = 0; j < 4; ++j)
            bf4[j] = frag_ld(frag_addr(Bs[cur], 32, (ng * 4 + j) * 16, 0, lane));
        #pragma unroll
        for (int j = 0; j < 4; ++j)
            c[j] = wmma_bf16(a, bf4[j], c[j]);
        __syncthreads();
    }
    #pragma unroll
    for (int j = 0; j < 4; ++j) {
        int ncol = n0 + (ng * 4 + j) * 16 + nl;
        #pragma unroll
        for (int r = 0; r < 8; ++r) {
            int m  = mi * 16 + r + 8 * half;
            int rg = row0 + m;                   // = b*T + t
            int b  = rg >> 10, t = rg & (T_SZ - 1);
            xin[((size_t)t * B_SZ + b) * H_SZ + ncol] = c[j][r];
        }
    }
}

// ---------------------------------------------------------------------------
// Kernel 2: persistent cooperative scan. 8 WGs x 512 thr.
// Resident W_recT slice in LDS (async DMA once); per step: async DMA bf16 h
// mirror -> LDS, prefetch f32 xin/h under the DMA, 16 pipelined WMMAs,
// tanh/blend f32, write hidden_states[t+1] + next bf16 mirror, grid barrier.
// ---------------------------------------------------------------------------
__global__ void __launch_bounds__(512)
k_scan(const float* __restrict__ xin, const unsigned short* __restrict__ WrecT,
       unsigned short* __restrict__ hbf,   // 2 x [B*H] bf16 mirrors (ping-pong)
       float* __restrict__ hid, unsigned int* __restrict__ bar) {
    extern __shared__ __align__(16) unsigned short sm[];
    unsigned short* hsA = sm;                 // 64 x 512 bf16 (h, staged per step)
    unsigned short* Bs  = sm + 64 * 512;      // 64 x 512 bf16 (W_recT, resident)

    const int tid = threadIdx.x;
    const int n0  = blockIdx.x * 64;
    for (int i = tid * 8; i < 64 * H_SZ; i += 512 * 8)     // one-time DMA
        async_ld_b128(&Bs[i], WrecT + (size_t)n0 * H_SZ + i);

    const int w = tid >> 5, lane = tid & 31;
    const int mi = w & 3, ni = w >> 2;        // 16 waves -> 4x4 16x16 tiles
    const int half = lane >> 4, nl = lane & 15;
    const int ng = n0 + ni * 16 + nl;         // this lane's global H column

    for (int t = 0; t < T_SZ; ++t) {
        const float* hcur = hid + (size_t)t * (B_SZ * H_SZ);
        float*       hnxt = hid + (size_t)(t + 1) * (B_SZ * H_SZ);
        const unsigned short* hsrc = hbf + (size_t)(t & 1) * (B_SZ * H_SZ);
        unsigned short*       hdst = hbf + (size_t)((t + 1) & 1) * (B_SZ * H_SZ);
        const float* xt = xin + (size_t)t * (B_SZ * H_SZ);

        for (int i = tid * 8; i < B_SZ * H_SZ; i += 512 * 8)   // 8 async b128
            async_ld_b128(&hsA[i], hsrc + i);

        // prefetch f32 epilogue operands under the DMA + WMMA chain
        float xtv[8], hcv[8];
        #pragma unroll
        for (int r = 0; r < 8; ++r) {
            int m = mi * 16 + r + 8 * half;
            xtv[r] = xt[(size_t)m * H_SZ + ng];
            hcv[r] = hcur[(size_t)m * H_SZ + ng];
        }

        wait_async0();
        __syncthreads();

        // software-pipelined 16-step WMMA chain
        v16bf a0 = frag_ld(frag_addr(hsA, H_SZ, mi * 16, 0, lane));
        v16bf b0 = frag_ld(frag_addr(Bs,  H_SZ, ni * 16, 0, lane));
        v8f c = {};
        #pragma unroll
        for (int ks = 32; ks < H_SZ; ks += 32) {
            v16bf a1 = frag_ld(frag_addr(hsA, H_SZ, mi * 16, ks, lane));
            v16bf b1 = frag_ld(frag_addr(Bs,  H_SZ, ni * 16, ks, lane));
            c = wmma_bf16(a0, b0, c);
            a0 = a1; b0 = b1;
        }
        c = wmma_bf16(a0, b0, c);

        #pragma unroll
        for (int r = 0; r < 8; ++r) {
            int m = mi * 16 + r + 8 * half;    // batch row
            float s = c[r] + xtv[r];
            s = tanhf(s);
            float hn = (1.0f - ALPHA_F) * hcv[r] + ALPHA_F * s;
            hnxt[(size_t)m * H_SZ + ng] = hn;       // == hidden_states[t+1]
            hdst[(size_t)m * H_SZ + ng] = f2bf(hn); // bf16 mirror for next DMA
        }

        // device-wide step barrier (monotonic counter, release/acquire fences)
        __syncthreads();
        if (tid == 0) {
            __threadfence();
            atomicAdd(bar, 1u);
            unsigned int target = (unsigned int)(NWG_SCAN * (t + 1));
            while (atomicAdd(bar, 0u) < target) __builtin_amdgcn_s_sleep(2);
            __threadfence();
        }
        __syncthreads();
    }
}

// ---------------------------------------------------------------------------
// Kernel 3: outputs[b][t][:] = hidden_states[t+1][b][:] @ W_out
// ---------------------------------------------------------------------------
__global__ void __launch_bounds__(256)
k_out_proj(const float* __restrict__ hidA, const float* __restrict__ Wout,
           float* __restrict__ out) {
    __shared__ __align__(16) unsigned short As[2][64 * 32];
    __shared__ __align__(16) unsigned short Bs[2][128 * 32];
    const int tid  = threadIdx.x;
    const int row0 = blockIdx.x * 64;
    const int n0   = blockIdx.y * 128;
    const int w = tid >> 5, lane = tid & 31;
    const int mi = w & 3, ng = w >> 2;
    const int half = lane >> 4, nl = lane & 15;
    const int NKB = H_SZ / 32;

    stage_A64x32(As[0], hidA, H_SZ, row0, 0, tid);
    stage_B128x32(Bs[0], Wout, O_SZ, 0, n0, tid);
    __syncthreads();

    v8f c[4] = {};
    for (int kb = 0; kb < NKB; ++kb) {
        int cur = kb & 1;
        if (kb + 1 < NKB) {
            stage_A64x32(As[cur ^ 1], hidA, H_SZ, row0, (kb + 1) * 32, tid);
            stage_B128x32(Bs[cur ^ 1], Wout, O_SZ, (kb + 1) * 32, n0, tid);
        }
        v16bf a = frag_ld(frag_addr(As[cur], 32, mi * 16, 0, lane));
        v16bf bf4[4];
        #pragma unroll
        for (int j = 0; j < 4; ++j)
            bf4[j] = frag_ld(frag_addr(Bs[cur], 32, (ng * 4 + j) * 16, 0, lane));
        #pragma unroll
        for (int j = 0; j < 4; ++j)
            c[j] = wmma_bf16(a, bf4[j], c[j]);
        __syncthreads();
    }
    #pragma unroll
    for (int j = 0; j < 4; ++j) {
        int ncol = n0 + (ng * 4 + j) * 16 + nl;
        #pragma unroll
        for (int r = 0; r < 8; ++r) {
            int m  = mi * 16 + r + 8 * half;
            int rg = row0 + m;                 // = t*B + b
            int t  = rg >> 6, b = rg & (B_SZ - 1);
            out[((size_t)b * T_SZ + t) * O_SZ + ncol] = c[j][r];
        }
    }
}

// ---------------------------------------------------------------------------
// Host launch
// ---------------------------------------------------------------------------
extern "C" void kernel_launch(void* const* d_in, const int* in_sizes, int n_in,
                              void* d_out, int out_size, void* d_ws, size_t ws_size,
                              hipStream_t stream) {
    (void)in_sizes; (void)n_in; (void)out_size; (void)ws_size;
    const float* x    = (const float*)d_in[0];   // [B,T,I]
    const float* h0   = (const float*)d_in[1];   // [B,H]
    const float* Win  = (const float*)d_in[2];   // [I,H]
    const float* Wrec = (const float*)d_in[3];   // [H,H]
    const float* Wout = (const float*)d_in[4];   // [H,O]

    float* out = (float*)d_out;                              // outputs [B,T,O]
    float* hid = out + (size_t)B_SZ * T_SZ * O_SZ;           // hidden [T+1,B,H]

    // workspace layout
    char* ws = (char*)d_ws;
    float*          xin   = (float*)ws;                      // [T,B,H] f32 (128 MB)
    size_t off = (size_t)T_SZ * B_SZ * H_SZ * sizeof(float);
    unsigned short* WrecT = (unsigned short*)(ws + off);     // [H,H] bf16 transposed
    off += (size_t)H_SZ * H_SZ * sizeof(unsigned short);
    unsigned short* hbf   = (unsigned short*)(ws + off);     // 2 x [B,H] bf16 mirror
    off += (size_t)2 * B_SZ * H_SZ * sizeof(unsigned short);
    unsigned int*   bar   = (unsigned int*)(ws + off);       // barrier counter

    // hidden_states[0] = h ; barrier counter = 0
    hipMemcpyAsync(hid, h0, (size_t)B_SZ * H_SZ * sizeof(float),
                   hipMemcpyDeviceToDevice, stream);
    hipMemsetAsync(bar, 0, 128, stream);

    k_prep_wrec<<<(H_SZ * H_SZ) / 256, 256, 0, stream>>>(Wrec, WrecT);
    k_prep_h<<<(B_SZ * H_SZ) / 256, 256, 0, stream>>>(h0, hbf);

    k_in_proj<<<dim3((B_SZ * T_SZ) / 64, H_SZ / 128), 256, 0, stream>>>(x, Win, xin);

    hipFuncSetAttribute(reinterpret_cast<const void*>(k_scan),
                        hipFuncAttributeMaxDynamicSharedMemorySize, 131072);
    k_scan<<<dim3(NWG_SCAN), dim3(512), 131072, stream>>>(xin, WrecT, hbf, hid, bar);

    k_out_proj<<<dim3((B_SZ * T_SZ) / 64, O_SZ / 128), 256, 0, stream>>>(
        hid + (size_t)B_SZ * H_SZ, Wout, out);
}
